// MobiusGRU_d_38190849196696
// MI455X (gfx1250) — matrix operand order repaired and checked by reading, not compiled, not run on hardware
//
#include <hip/hip_runtime.h>
#include <hip/hip_bf16.h>
#include <stdint.h>

// ---------------------------------------------------------------------------
// Hyperbolic (Lorentz) GRU for MI455X / gfx1250.
//   K0: pack weights f32 -> f16, zero-padded [3][288][288]
//   K1: pack inputs  [0, x_t] -> f16 panels Xin[col=t*B+b][288]
//   K2: time-parallel input projection GEMM (WMMA f16->f32), TDM-staged B panel
//       Xproj[col][3*288] (f16, 56.6 MB -> L2 resident on MI455X's 192MB L2)
//   K3: persistent single-WGP recurrent kernel, all state in 288KB LDS
//       (fits CDNA5's 320KB/WGP), WMMA gate GEMMs + wave32 hyperbolic math
// ---------------------------------------------------------------------------

typedef __attribute__((ext_vector_type(16))) _Float16 v16h;
typedef __attribute__((ext_vector_type(8)))  _Float16 v8h;
typedef __attribute__((ext_vector_type(8)))  float    v8f;
typedef __attribute__((ext_vector_type(4)))  unsigned int u32x4;
typedef __attribute__((ext_vector_type(4)))  int      i32x4;
typedef __attribute__((ext_vector_type(8)))  int      i32x8;

#define B_    64
#define T_    512
#define DIN   256
#define H1    257          // hidden+time component
#define P     288          // padded dim (multiple of 32 for K, 16 for M/N)
#define P3    (3*P)        // 864
#define NCOL  (T_*B_)      // 32768 time*batch columns

#define EPSF    1e-7f
#define MINNORM 1e-15f
#define MAXNORM 1000000.0f

// workspace layout (bytes); total ~81 MB
#define WS_WI    ((size_t)0)          // 3*288*288 f16 = 486 KB
#define WS_WH    ((size_t)1 << 20)    // 3*288*288 f16
#define WS_XIN   ((size_t)2 << 20)    // 32768*288 f16 = 18.9 MB
#define WS_XPROJ ((size_t)24 << 20)   // 32768*864 f16 = 56.6 MB

__device__ __forceinline__ float wave_sum(float v) {
  // wave32 butterfly reduction
  #pragma unroll
  for (int off = 16; off > 0; off >>= 1) v += __shfl_xor(v, off, 32);
  return v;
}
__device__ __forceinline__ float sigmoidf(float x) { return 1.0f / (1.0f + __expf(-x)); }

// One 16x16 output tile, K=288 via 9x v_wmma_f32_16x16x32_f16.
// Two interleaved accumulators: consecutive WMMAs are independent, so the
// f16 WMMA->WMMA D->A/B hazard window is filled by the A-panel loads and
// the sibling accumulator's WMMA instead of v_nops.
// Arow: base of A row (M = mBase + lane&15), row stride already applied.
// Bcol: base of B column (N = nBase + lane&15), column-contiguous in K.
__device__ __forceinline__ v8f wmma_tile_288(const _Float16* __restrict__ Arow,
                                             const _Float16* __restrict__ Bcol,
                                             int lane) {
  const int half = lane >> 4;       // 0: lanes 0-15, 1: lanes 16-31
  v8f c0 = {0.f, 0.f, 0.f, 0.f, 0.f, 0.f, 0.f, 0.f};
  v8f c1 = {0.f, 0.f, 0.f, 0.f, 0.f, 0.f, 0.f, 0.f};
  #pragma unroll
  for (int k0 = 0; k0 < P; k0 += 32) {
    // A fragment: lane<16 holds K = k0+{0..15},
    // lane>=16 holds K = k0+{16..31}  -> two 16B chunks (matches B)
    union { v16h v; struct { v8h lo, hi; } p; } af;
    af.p.lo = *(const v8h*)(Arow + k0 + 16 * half);
    af.p.hi = *(const v8h*)(Arow + k0 + 16 * half + 8);
    // B fragment: lane&15 selects N column, half-wave selects K 16-block
    v16h bf = *(const v16h*)(Bcol + k0 + 16 * half);
    if (((k0 >> 5) & 1) == 0)
      c0 = __builtin_amdgcn_wmma_f32_16x16x32_f16(false, af.v, false, bf,
                                                  (short)0, c0, false, false);
    else
      c1 = __builtin_amdgcn_wmma_f32_16x16x32_f16(false, af.v, false, bf,
                                                  (short)0, c1, false, false);
  }
  return c0 + c1;
}

// -------------------------------- K0 ---------------------------------------
__global__ void pack_weights_kernel(const float* __restrict__ wih,
                                    const float* __restrict__ whh,
                                    _Float16* __restrict__ WiH,
                                    _Float16* __restrict__ WhH) {
  int idx = blockIdx.x * blockDim.x + threadIdx.x;
  if (idx >= 3 * P * P) return;
  int g = idx / (P * P);
  int r = idx % (P * P);
  int m = r / P, kcol = r % P;
  float vi = 0.f, vh = 0.f;
  if (m < H1 && kcol < H1) {
    vi = wih[(size_t)(g * H1 + m) * H1 + kcol];
    vh = whh[(size_t)(g * H1 + m) * H1 + kcol];
  }
  WiH[idx] = (_Float16)vi;
  WhH[idx] = (_Float16)vh;
}

// -------------------------------- K1 ---------------------------------------
// Xin[col][k]: k=0 -> 0 (time component of tangent), k in [1,256] -> x[b][t][k-1]
__global__ void pack_x_kernel(const float* __restrict__ x,
                              _Float16* __restrict__ XinH) {
  size_t idx = (size_t)blockIdx.x * blockDim.x + threadIdx.x;
  if (idx >= (size_t)NCOL * P) return;
  int col = (int)(idx / P);
  int k   = (int)(idx % P);
  int t = col / B_, b = col % B_;
  float v = 0.f;
  if (k >= 1 && k <= DIN) v = x[((size_t)b * T_ + t) * DIN + (k - 1)];
  XinH[idx] = (_Float16)v;
}

// -------------------------------- K2 ---------------------------------------
// Xproj[col][g*288+m] = sum_k Wi[g][m][k] * Xin[col][k]  + bias_tangent[g][m]
// One block per 16 columns; TDM stages the 16x288 f16 B panel into LDS.
__global__ void input_proj_kernel(const _Float16* __restrict__ WiH,
                                  const _Float16* __restrict__ XinH,
                                  const float* __restrict__ bias,
                                  _Float16* __restrict__ XprojH) {
  __shared__ _Float16 Bp[16 * P];   // 9 KB
  const int colBase = blockIdx.x * 16;

#if __has_builtin(__builtin_amdgcn_tensor_load_to_lds)
  // Real LDS byte address of Bp: goes into the D# lds_addr field AND makes
  // Bp's address escape, so alias analysis must assume the TDM writes it
  // (otherwise the B-fragment ds_loads below get folded to undef).
  unsigned ldsAddr = (unsigned)(uintptr_t)(&Bp[0]);
  if ((threadIdx.x >> 5) == 0) {   // one wave issues the TDM descriptor
    unsigned long long ga =
        (unsigned long long)(uintptr_t)(XinH + (size_t)colBase * P);
    u32x4 g0;
    g0.x = 1u;                                   // count=1 user descriptor
    g0.y = ldsAddr;                              // lds_addr (bytes)
    g0.z = (unsigned)(ga & 0xffffffffu);         // global_addr[31:0]
    g0.w = (unsigned)((ga >> 32) & 0x01ffffffu)  // global_addr[56:32]
         | (2u << 30);                           // type=2 ("image")
    i32x8 g1;
    g1[0] = 0x00010000;                          // data_size=2B
    g1[1] = (int)((P & 0xffff) << 16);           // tensor_dim0 = 288
    g1[2] = (int)0x80000000;                     // tensor_dim1 = 32768 (lo16)
    g1[3] = (int)(P << 16);                      // tile_dim0 = 288
    g1[4] = 16;                                  // tile_dim1 = 16 rows
    g1[5] = P;                                   // tensor_dim0_stride = 288
    g1[6] = 0;
    g1[7] = 0;
    i32x4 z4 = {0, 0, 0, 0};
#if defined(__clang_major__) && (__clang_major__ >= 23)
    i32x8 z8 = {0, 0, 0, 0, 0, 0, 0, 0};
    __builtin_amdgcn_tensor_load_to_lds(g0, g1, z4, z4, z8, 0);
#else
    __builtin_amdgcn_tensor_load_to_lds(g0, g1, z4, z4, 0);
#endif
    __builtin_amdgcn_s_wait_tensorcnt(0);
  }
  // Compiler-level fence: with Bp address-taken above, this forces the
  // assumption that Bp's contents changed before the GEMM reads it.
  asm volatile("" ::: "memory");
#else
  for (int idx = threadIdx.x; idx < 16 * P; idx += blockDim.x)
    Bp[idx] = XinH[(size_t)colBase * P + idx];
#endif
  __syncthreads();

  const int lane = threadIdx.x & 31;
  const int wv   = threadIdx.x >> 5;   // 8 waves
  const int l16  = lane & 15;
  const int half = lane >> 4;

  // 54 row tiles (3 gates x 18), round-robin over 8 waves
  for (int rt = wv; rt < 54; rt += 8) {
    int g = rt / 18;
    int mBase = (rt % 18) * 16;
    const _Float16* Arow = WiH + ((size_t)g * P + mBase + l16) * P;
    const _Float16* Bcol = Bp + (size_t)l16 * P;
    v8f c = wmma_tile_288(Arow, Bcol, lane);
    // C/D layout: VGPR v -> M = mBase + v + 8*half ; N = l16
    int col = colBase + l16;
    _Float16* orow = XprojH + (size_t)col * P3 + (size_t)g * P;
    #pragma unroll
    for (int v = 0; v < 8; ++v) {
      int m = mBase + v + 8 * half;
      float bv = (m < H1) ? bias[g * H1 + m] : 0.f;  // expmap0(b) tangent
      orow[m] = (_Float16)(c[v] + bv);
    }
  }
}

// -------------------------------- K3 ---------------------------------------
// Persistent recurrent kernel: 1 block, 512 threads (16 wave32), 288 KB LDS.
__global__ void recurrent_kernel(const _Float16* __restrict__ WhH,
                                 const _Float16* __restrict__ XprojH,
                                 const float* __restrict__ kptr,
                                 float* __restrict__ out) {
  extern __shared__ unsigned char smemRaw[];
  float*    Hs   = (float*)smemRaw;          // [64][288] hyperbolic state
  float*    GzB  = Hs  + B_ * P;             // [64][288] gate z (then sigmoid)
  float*    GrB  = GzB + B_ * P;             // [64][288] gate r, then h_tilde
  _Float16* hlog = (_Float16*)(GrB + B_ * P);// [64][288] logmap0(h), f16
  _Float16* rhlg = hlog + B_ * P;            // [64][288] r * logmap0(h), f16

  const float kk    = kptr[0];
  const float Kv    = 1.0f / kk;
  const float sqrtK = sqrtf(Kv);

  const int lane = threadIdx.x & 31;
  const int wv   = threadIdx.x >> 5;   // 16 waves
  const int l16  = lane & 15;
  const int half = lane >> 4;

  // h0 = 0 (reference init)
  for (int idx = threadIdx.x; idx < B_ * P; idx += blockDim.x) Hs[idx] = 0.f;
  __syncthreads();

  for (int t = 0; t < T_; ++t) {
    // prefetch next step's Xproj panel (L2-resident; emits global_prefetch_b8)
    if (t + 1 < T_)
      __builtin_prefetch(XprojH + (size_t)(t + 1) * B_ * P3 +
                         (size_t)threadIdx.x * 96, 0, 1);

    // ---- phase 1: hlog = logmap0(h)  (4 batch rows per wave) ----
    for (int i = 0; i < 4; ++i) {
      int b = wv + 16 * i;
      float s = 0.f;
      #pragma unroll
      for (int j = 0; j < 9; ++j) {
        int kx = lane + 32 * j;
        float hv = Hs[b * P + kx];
        if (kx >= 1) s += hv * hv;       // padded region already 0
      }
      s = wave_sum(s);
      float yn = fmaxf(sqrtf(s), MINNORM);
      float th = fmaxf(Hs[b * P] / sqrtK, 1.0f + EPSF);
      float coef = sqrtK * acoshf(th) / yn;
      #pragma unroll
      for (int j = 0; j < 9; ++j) {
        int kx = lane + 32 * j;
        float v = (kx == 0 || kx >= H1) ? 0.f : coef * Hs[b * P + kx];
        hlog[b * P + kx] = (_Float16)v;
      }
    }
    __syncthreads();

    // ---- phase 2: gate GEMMs  Gr = W_hr@hlog + Xproj_r, Gz = W_hz@hlog + Xproj_z
    for (int tl = wv; tl < 144; tl += 16) {
      int which = tl / 72;               // 0 -> r (g=0), 1 -> z (g=2)
      int g = which ? 2 : 0;
      float* Gout = which ? GzB : GrB;
      int t72 = tl % 72;
      int mBase = (t72 >> 2) * 16;
      int nBase = (t72 & 3) * 16;
      const _Float16* Arow = WhH + ((size_t)g * P + mBase + l16) * P;
      const _Float16* Bcol = hlog + (size_t)(nBase + l16) * P;
      v8f c = wmma_tile_288(Arow, Bcol, lane);
      int b = nBase + l16;
      const _Float16* xp = XprojH + ((size_t)t * B_ + b) * P3 + (size_t)g * P;
      #pragma unroll
      for (int v = 0; v < 8; ++v) {
        int m = mBase + v + 8 * half;
        Gout[b * P + m] = c[v] + (float)xp[m];
      }
    }
    __syncthreads();

    // ---- phase 3: z = sigmoid(Gz); rh_log = sigmoid(Gr) * hlog ----
    for (int idx = threadIdx.x; idx < B_ * P; idx += blockDim.x) {
      float z = sigmoidf(GzB[idx]);
      float r = sigmoidf(GrB[idx]);
      GzB[idx] = z;
      int kx = idx % P;
      float rh = (kx == 0 || kx >= H1) ? 0.f : r * (float)hlog[idx];
      rhlg[idx] = (_Float16)rh;
    }
    __syncthreads();

    // ---- phase 4: h_tilde tangent = W_hh @ rh_log + Xproj_h (into GrB) ----
    for (int tl = wv; tl < 72; tl += 16) {
      int mBase = (tl >> 2) * 16;
      int nBase = (tl & 3) * 16;
      const _Float16* Arow = WhH + ((size_t)1 * P + mBase + l16) * P;
      const _Float16* Bcol = rhlg + (size_t)(nBase + l16) * P;
      v8f c = wmma_tile_288(Arow, Bcol, lane);
      int b = nBase + l16;
      const _Float16* xp = XprojH + ((size_t)t * B_ + b) * P3 + (size_t)1 * P;
      #pragma unroll
      for (int v = 0; v < 8; ++v) {
        int m = mBase + v + 8 * half;
        GrB[b * P + m] = c[v] + (float)xp[m];
      }
    }
    __syncthreads();

    // ---- phase 5: h <- expmap(ptransp0(h, z*(ht_tan - hlog)), h), proj ----
    for (int i = 0; i < 4; ++i) {
      int b = wv + 16 * i;
      float hr[9], vr[9], ur[9];
      float ssy = 0.f;
      #pragma unroll
      for (int j = 0; j < 9; ++j) {
        int kx = lane + 32 * j;
        float hv = Hs[b * P + kx];
        hr[j] = hv;
        float vv = GzB[b * P + kx] *
                   (GrB[b * P + kx] - (float)hlog[b * P + kx]);
        if (kx == 0 || kx >= H1) vv = 0.f;
        vr[j] = vv;
        if (kx >= 1) ssy += hv * hv;
      }
      ssy = wave_sum(ssy);
      float yn = fmaxf(sqrtf(ssy), MINNORM);
      float x0 = __shfl(hr[0], 0, 32);          // h[b][0] lives in lane 0, j=0

      // ptransp0: alpha = <y/yn, v[1:]> / sqrtK ; u = v - alpha * [-yn, (sqrtK-x0)*y/yn]
      float sa = 0.f;
      #pragma unroll
      for (int j = 0; j < 9; ++j) {
        int kx = lane + 32 * j;
        if (kx >= 1) sa += hr[j] * vr[j];
      }
      sa = wave_sum(sa);
      float alpha = sa / (yn * sqrtK);
      #pragma unroll
      for (int j = 0; j < 9; ++j) {
        int kx = lane + 32 * j;
        ur[j] = (kx == 0) ? (vr[j] + alpha * yn)
                          : (vr[j] - alpha * (sqrtK - x0) * hr[j] / yn);
      }
      // proj_tan: u0 = <h[1:], u[1:]> / max(x0, EPS)
      float su = 0.f;
      #pragma unroll
      for (int j = 0; j < 9; ++j) {
        int kx = lane + 32 * j;
        if (kx >= 1) su += hr[j] * ur[j];
      }
      su = wave_sum(su);
      float u0 = su / fmaxf(x0, EPSF);

      // expmap(w, h): theta = clip(mink_norm(w))/sqrtK
      float sw = 0.f;
      #pragma unroll
      for (int j = 0; j < 9; ++j) {
        int kx = lane + 32 * j;
        if (kx >= 1) sw += ur[j] * ur[j];
      }
      sw = wave_sum(sw);
      float md = sw - u0 * u0;                    // sum(w*w) - 2*w0^2
      float nrm = fminf(sqrtf(fmaxf(md, EPSF)), MAXNORM);
      float theta = fmaxf(nrm / sqrtK, MINNORM);
      float ch = coshf(theta), shc = sinhf(theta) / theta;

      float s2 = 0.f, hnr[9];
      #pragma unroll
      for (int j = 0; j < 9; ++j) {
        int kx = lane + 32 * j;
        float w = (kx == 0) ? u0 : ur[j];
        float hn = ch * hr[j] + shc * w;
        if (kx >= H1) hn = 0.f;
        hnr[j] = hn;
        if (kx >= 1) s2 += hn * hn;
      }
      s2 = wave_sum(s2);
      float h0n = sqrtf(fmaxf(Kv + s2, EPSF));    // proj(): rebuild time comp

      #pragma unroll
      for (int j = 0; j < 9; ++j) {
        int kx = lane + 32 * j;
        float o = (kx == 0) ? h0n : hnr[j];
        if (kx < H1) {
          Hs[b * P + kx] = o;
          out[(size_t)t * B_ * H1 + (size_t)b * H1 + kx] = o;   // outs[t]
        } else {
          Hs[b * P + kx] = 0.f;
        }
      }
    }
    __syncthreads();
  }

  // ht = h_last[..., 1:]  (appended after outs)
  for (int idx = threadIdx.x; idx < B_ * DIN; idx += blockDim.x) {
    int b = idx / DIN, kx = idx % DIN + 1;
    out[(size_t)T_ * B_ * H1 + idx] = Hs[b * P + kx];
  }
}

// ---------------------------------------------------------------------------
extern "C" void kernel_launch(void* const* d_in, const int* in_sizes, int n_in,
                              void* d_out, int out_size, void* d_ws, size_t ws_size,
                              hipStream_t stream) {
  const float* x    = (const float*)d_in[0];
  const float* kp   = (const float*)d_in[1];
  const float* wih  = (const float*)d_in[2];
  const float* whh  = (const float*)d_in[3];
  const float* bias = (const float*)d_in[4];
  float* out = (float*)d_out;

  uint8_t* ws = (uint8_t*)d_ws;
  _Float16* WiH    = (_Float16*)(ws + WS_WI);
  _Float16* WhH    = (_Float16*)(ws + WS_WH);
  _Float16* XinH   = (_Float16*)(ws + WS_XIN);
  _Float16* XprojH = (_Float16*)(ws + WS_XPROJ);

  {
    int n = 3 * P * P;
    pack_weights_kernel<<<(n + 255) / 256, 256, 0, stream>>>(wih, whh, WiH, WhH);
  }
  {
    size_t n = (size_t)NCOL * P;
    pack_x_kernel<<<(unsigned)((n + 255) / 256), 256, 0, stream>>>(x, XinH);
  }
  input_proj_kernel<<<NCOL / 16, 256, 0, stream>>>(WiH, XinH, bias, XprojH);

  const size_t smem = (size_t)(3 * B_ * P) * sizeof(float)
                    + (size_t)(2 * B_ * P) * sizeof(_Float16); // 294912 B
  recurrent_kernel<<<1, 512, smem, stream>>>(WhH, XprojH, kp, out);
}
